// MnistSNN_Exact_50156628082698
// MI455X (gfx1250) — matrix operand order repaired and checked by reading, not compiled, not run on hardware
//
#include <hip/hip_runtime.h>
#include <hip/hip_bf16.h>

typedef __bf16 v16bf __attribute__((ext_vector_type(16)));
typedef float  v8f   __attribute__((ext_vector_type(8)));

#define B_SZ    4096
#define TW      80
#define TC_IN_F 17.452274f

// ---- workspace layout (bytes) ----
#define OFF_ST    0                           // spike-in times: B*784 f32
#define SZ_ST     (B_SZ * 784 * 4)
#define OFF_F1    (OFF_ST + SZ_ST)            // layer1 fired: B*576 u16 (12 ch bits)
#define SZ_F1     (B_SZ * 576 * 2)
#define OFF_PM1   (OFF_F1 + SZ_F1)            // layer1 pooled cum mask: B*144 u16 (ch bits per pos)
#define SZ_PM1    (B_SZ * 144 * 2)
#define OFF_PM1T  (OFF_PM1 + SZ_PM1)          // transposed: B*[ic:12][y:12] u32, bit x
#define SZ_PM1T   (B_SZ * 144 * 4)
#define OFF_F2    (OFF_PM1T + SZ_PM1T)        // layer2 fired: B*64 pos * 2 halves u32
#define SZ_F2     (B_SZ * 64 * 2 * 4)
#define OFF_PM2   (OFF_F2 + SZ_F2)            // layer2 pooled cum mask: B*32 u32 (NHWC bit order)
#define SZ_PM2    (B_SZ * 32 * 4)
#define OFF_F3    (OFF_PM2 + SZ_PM2)          // fc fired: B u32 (10 bits)
#define SZ_F3     (B_SZ * 4)
#define OFF_W2B   (OFF_F3 + SZ_F3)            // conv2 weights, bf16 3-split: [sel][kb][ntl][lane][s][e]
#define SZ_W2B    (2 * 10 * 2 * 32 * 3 * 16 * 2)   // = 122880 B
#define OFF_FWB   (OFF_W2B + SZ_W2B)          // fc weights, bf16 3-split: [kb][lane][s][e]
#define SZ_FWB    (32 * 32 * 3 * 16 * 2)      // = 98304 B
#define STATE_BYTES (OFF_W2B - OFF_F1)

// bf16 element index -> K offset within a 32-wide K block (ISA 16-bit A layout, mirrored for B)
__device__ __forceinline__ int kmap(int e, int h) { return (e < 8) ? (e + 8 * h) : (e + 8 + 8 * h); }

__device__ __forceinline__ unsigned short f2bf_rne(float f) {
  unsigned int u = __float_as_uint(f);
  return (unsigned short)((u + 0x7FFFu + ((u >> 16) & 1u)) >> 16);
}
__device__ __forceinline__ float bf2f(unsigned short b) {
  return __uint_as_float(((unsigned int)b) << 16);
}
__device__ __forceinline__ unsigned short bf_split(float val, int s) {
  unsigned short b0 = f2bf_rne(val);
  if (s == 0) return b0;
  float r1 = val - bf2f(b0);
  unsigned short b1 = f2bf_rne(r1);
  if (s == 1) return b1;
  return f2bf_rne(r1 - bf2f(b1));
}

#define WMMA_BF16(A, Bv, C) \
  __builtin_amdgcn_wmma_f32_16x16x32_bf16(false, (A), false, (Bv), (short)0, (C), false, false)

// -------- prep kernels --------
__global__ __launch_bounds__(256) void k_spike_times(const float* __restrict__ img,
                                                     float* __restrict__ st, int n) {
  int i = blockIdx.x * 256 + threadIdx.x;
  if (i < n) {
    float p = fmaxf(img[i], 1e-5f);
    st[i] = ceilf(fmaxf(-TC_IN_F * logf(p), 0.0f));
  }
}

__global__ __launch_bounds__(256) void k_init_out(float* __restrict__ out, int n) {
  int i = blockIdx.x * 256 + threadIdx.x;
  if (i < n) out[i] = 9999.0f;
}

// conv2 weights -> bf16 3-split, layout [sel][kb][ntl][lane][s][e].  61440 threads.
__global__ __launch_bounds__(256) void k_prep_w2(const float* __restrict__ w2,  // 64*300
                                                 unsigned short* __restrict__ w2b) {
  const int idx = blockIdx.x * 256 + threadIdx.x;
  const int e = idx & 15;
  const int r = idx >> 4;
  const int s = r % 3;
  const int r2 = r / 3;
  const int lane = r2 & 31;
  const int r3 = r2 >> 5;
  const int ntl = r3 & 1;
  const int r4 = r3 >> 1;
  const int kb = r4 % 10, sel = r4 / 10;
  const int l = lane & 15, h = lane >> 4;
  const int k = kb * 32 + kmap(e, h);
  const int c = sel * 32 + ntl * 16 + l;
  const float val = (k < 300) ? w2[c * 300 + k] : 0.0f;
  w2b[idx] = bf_split(val, s);
}

// fc weights -> bf16 3-split, layout [kb][lane][s][e] (N padded 10->16).  49152 threads.
__global__ __launch_bounds__(256) void k_prep_fw(const float* __restrict__ fw,  // 1024*10
                                                 unsigned short* __restrict__ fwb) {
  const int idx = blockIdx.x * 256 + threadIdx.x;
  const int e = idx & 15;
  const int r = idx >> 4;
  const int s = r % 3;
  const int r2 = r / 3;
  const int lane = r2 & 31;
  const int kb = r2 >> 5;
  const int l = lane & 15, h = lane >> 4;
  const int k = kb * 32 + kmap(e, h);
  const float val = (l < 10) ? fw[k * 10 + l] : 0.0f;
  fwb[idx] = bf_split(val, s);
}

// -------- layer 1: conv1 + neuron + ttfs-pool (direct, 25 taps) --------
__global__ __launch_bounds__(256) void k_layer1(const float* __restrict__ st,
                                                const float* __restrict__ w1,   // 12*25
                                                const float* __restrict__ b1,   // 12
                                                unsigned short* __restrict__ f1,
                                                unsigned short* __restrict__ pm1,
                                                unsigned int* __restrict__ pm1t,
                                                int t) {
  __shared__ float s_st[784];
  __shared__ float s_w[300];
  __shared__ float s_b[12];
  __shared__ unsigned int s_spk[576];
  __shared__ unsigned int s_pool[144];
  const int b = blockIdx.x, tid = threadIdx.x;
  for (int i = tid; i < 784; i += 256) s_st[i] = st[b * 784 + i];
  for (int i = tid; i < 300; i += 256) s_w[i] = w1[i];
  if (tid < 12) s_b[tid] = b1[tid];
  __syncthreads();

  const float tf = (float)t;
  const float vth = expf(-tf * 0.05f);
  const float vgate = (vth > 1e-5f) ? vth : 3.0e38f;
  const float bs = tf + 1.0f;

  for (int pos = tid; pos < 576; pos += 256) {
    const int y = pos / 24, x = pos % 24;
    float ind[25];
#pragma unroll
    for (int i = 0; i < 5; ++i)
#pragma unroll
      for (int j = 0; j < 5; ++j)
        ind[i * 5 + j] = (s_st[(y + i) * 28 + (x + j)] <= tf + 0.5f) ? 1.0f : 0.0f;
    const unsigned int fired_old = f1[b * 576 + pos];
    unsigned int fire_bits = 0;
#pragma unroll 1
    for (int c = 0; c < 12; ++c) {
      float v = 0.0f;
#pragma unroll
      for (int k = 0; k < 25; ++k) v = fmaf(s_w[c * 25 + k], ind[k], v);
      v += bs * s_b[c];
      const bool fire = (v >= vgate) && !((fired_old >> c) & 1u);
      fire_bits |= (fire ? 1u : 0u) << c;
    }
    f1[b * 576 + pos] = (unsigned short)(fired_old | fire_bits);
    s_spk[pos] = fire_bits;
  }
  __syncthreads();
  for (int pp = tid; pp < 144; pp += 256) {
    const int py = pp / 12, px = pp % 12;
    const unsigned int p = s_spk[(2 * py) * 24 + 2 * px] | s_spk[(2 * py) * 24 + 2 * px + 1] |
                           s_spk[(2 * py + 1) * 24 + 2 * px] | s_spk[(2 * py + 1) * 24 + 2 * px + 1];
    const unsigned int old = pm1[b * 144 + pp];
    const unsigned int nw = old | (p & ~old & 0xFFFu);
    pm1[b * 144 + pp] = (unsigned short)nw;
    s_pool[pp] = nw;
  }
  __syncthreads();
  for (int i = tid; i < 144; i += 256) {     // transpose: per-channel row bitmasks
    const int ic = i / 12, py = i % 12;
    unsigned int bits = 0;
#pragma unroll
    for (int px = 0; px < 12; ++px) bits |= ((s_pool[py * 12 + px] >> ic) & 1u) << px;
    pm1t[b * 144 + ic * 12 + py] = bits;
  }
}

// -------- layer 2: conv2 as bf16 WMMA GEMM (M=64 pos, N=32 ch/half, K=320, 3-split) --------
__global__ __launch_bounds__(256) void k_layer2(const unsigned int* __restrict__ pm1t,
                                                const uint4* __restrict__ w2b,   // swizzled frags
                                                const float* __restrict__ b2,
                                                unsigned int* __restrict__ f2,
                                                unsigned int* __restrict__ pm2,
                                                int t) {
  __shared__ uint4 sW2q[3840];          // 61440 B: [kb][ntl][lane][s] x 32B fragments
  __shared__ float sB[32];
  __shared__ unsigned int sInT[144];    // [ic:12][y:12], bit x
  __shared__ unsigned int sFire[64];    // 8x8 pos, 32 channel bits (this half)

  const int tid = threadIdx.x;
  const int sel = blockIdx.x & 1;
  const int lane = tid & 31, wave = tid >> 5;
  const int l = lane & 15, h = lane >> 4;
  const int mtile = wave >> 1;
  const int ntl = wave & 1;
  const int nl = ntl * 16 + l;          // C/B column role of l
  const int m = mtile * 16 + l;         // A row role of l
  const int oy = m >> 3, ox = m & 7;

  if (tid == 0) __builtin_prefetch((const void*)(w2b + sel * 1920), 0, 3);
  for (int i = tid; i < 3840; i += 256) sW2q[i] = w2b[sel * 3840 + i];
  if (tid < 32) sB[tid] = b2[sel * 32 + tid];
  __syncthreads();

  const float myb = sB[nl];             // hoisted bias for this lane's column
  const float tf = (float)t;
  const float vth = expf(-tf * 0.05f);
  const float vgate = (vth > 1e-5f) ? vth : 3.0e38f;
  const float bs = tf + 1.0f;
  const int img_stride = (int)(gridDim.x >> 1);
  const uint4* wbase = &sW2q[(ntl * 32 + lane) * 6];   // + kb*384

  for (int img = (int)(blockIdx.x >> 1); img < B_SZ; img += img_stride) {
    if (tid < 144) sInT[tid] = pm1t[img * 144 + tid];
    if (tid < 64) sFire[tid] = 0u;
    __syncthreads();

    // build this lane's 320-bit im2col row (k = ic*25 + ii*5 + jj), once per image
    unsigned int rowbits[10];
#pragma unroll
    for (int w = 0; w < 10; ++w) rowbits[w] = 0u;
#pragma unroll
    for (int ic = 0; ic < 12; ++ic) {
      unsigned int b25 = 0;
#pragma unroll
      for (int ii = 0; ii < 5; ++ii)
        b25 |= ((sInT[ic * 12 + oy + ii] >> ox) & 0x1Fu) << (5 * ii);
      const int off = ic * 25, w = off >> 5, sh = off & 31;
      rowbits[w] |= b25 << sh;
      if (sh > 7) rowbits[w + 1] |= b25 >> (32 - sh);
    }

    v8f acc = {};
#pragma unroll 1
    for (int kb = 0; kb < 10; ++kb) {
      const unsigned int bits = rowbits[kb];
      union { unsigned int u[8]; v16bf v; } A;
#pragma unroll
      for (int i = 0; i < 8; ++i) {
        const int k0 = (i < 4) ? (2 * i + 8 * h) : (2 * i + 8 + 8 * h);
        A.u[i] = (((bits >> k0) & 1u) ? 0x3F80u : 0u) |
                 (((bits >> (k0 + 1)) & 1u) ? 0x3F800000u : 0u);
      }
      const uint4* wp = wbase + kb * 384;   // [kb][ntl][lane]: 3 splits x 32B contiguous
      union { uint4 q[2]; v16bf v; } B0, B1, B2;
      B0.q[0] = wp[0]; B0.q[1] = wp[1];
      B1.q[0] = wp[2]; B1.q[1] = wp[3];
      B2.q[0] = wp[4]; B2.q[1] = wp[5];
      acc = WMMA_BF16(A.v, B0.v, acc);
      acc = WMMA_BF16(A.v, B1.v, acc);
      acc = WMMA_BF16(A.v, B2.v, acc);
    }

    // fire bits via ballot -> LDS
#pragma unroll
    for (int v = 0; v < 8; ++v) {
      const unsigned int bal = __builtin_amdgcn_ballot_w32(acc[v] + bs * myb >= vgate);
      if (lane == 0) {
        atomicOr(&sFire[mtile * 16 + v], (bal & 0xFFFFu) << (16 * ntl));
        atomicOr(&sFire[mtile * 16 + v + 8], (bal >> 16) << (16 * ntl));
      }
    }
    __syncthreads();
    if (tid < 64) {
      const unsigned int raw = sFire[tid];
      const unsigned int gi = (unsigned int)(img * 64 + tid) * 2u + (unsigned int)sel;
      const unsigned int old = f2[gi];
      const unsigned int fire = raw & ~old;
      f2[gi] = old | fire;
      sFire[tid] = fire;                 // masked spikes feed pooling
    }
    __syncthreads();
    if (tid < 16) {
      const int py = tid >> 2, px = tid & 3;
      const int p00 = (2 * py) * 8 + 2 * px;
      const unsigned int pooled = sFire[p00] | sFire[p00 + 1] | sFire[p00 + 8] | sFire[p00 + 9];
      const unsigned int pidx = (unsigned int)img * 32u + (unsigned int)tid * 2u + (unsigned int)sel;
      const unsigned int old = pm2[pidx];
      pm2[pidx] = old | (pooled & ~old);
    }
    __syncthreads();
  }
}

// -------- fc: bf16 WMMA GEMM (M=16 batch, N=16 (10 used), K=1024, 3-split) --------
__global__ __launch_bounds__(256) void k_fc(const unsigned int* __restrict__ pm2,  // B*32 u32
                                            const uint4* __restrict__ fwb,         // swizzled frags
                                            const float* __restrict__ fb,
                                            unsigned int* __restrict__ f3,
                                            float* __restrict__ out, int t) {
  __shared__ float sB[16];
  const int tid = threadIdx.x;
  if (tid == 0) __builtin_prefetch((const void*)fwb, 0, 3);
  if (tid < 16) sB[tid] = (tid < 10) ? fb[tid] : 0.0f;
  __syncthreads();

  const int lane = tid & 31, wave = tid >> 5;
  const int l = lane & 15, h = lane >> 4;
  const int b0 = blockIdx.x * 128 + wave * 16;
  const float myb = sB[l];
  const float tf = (float)t;
  const float vth = expf(-tf * 0.05f);
  const float vgate = (vth > 1e-5f) ? vth : 3.0e38f;
  const float bs = tf + 1.0f;

  v8f acc = {};
  const unsigned int* rowp = pm2 + (b0 + l) * 32;
  const uint4* wbase = fwb + lane * 6;        // + kb*192
#pragma unroll 1
  for (int kb = 0; kb < 32; ++kb) {
    const unsigned int bits = rowp[kb];
    union { unsigned int u[8]; v16bf v; } A;
#pragma unroll
    for (int i = 0; i < 8; ++i) {
      const int k0 = (i < 4) ? (2 * i + 8 * h) : (2 * i + 8 + 8 * h);
      A.u[i] = (((bits >> k0) & 1u) ? 0x3F80u : 0u) |
               (((bits >> (k0 + 1)) & 1u) ? 0x3F800000u : 0u);
    }
    const uint4* wp = wbase + kb * 192;       // [kb][lane]: 3 splits x 32B contiguous
    union { uint4 q[2]; v16bf v; } B0, B1, B2;
    B0.q[0] = wp[0]; B0.q[1] = wp[1];
    B1.q[0] = wp[2]; B1.q[1] = wp[3];
    B2.q[0] = wp[4]; B2.q[1] = wp[5];
    acc = WMMA_BF16(A.v, B0.v, acc);
    acc = WMMA_BF16(A.v, B1.v, acc);
    acc = WMMA_BF16(A.v, B2.v, acc);
  }
  if (l < 10) {
#pragma unroll
    for (int v = 0; v < 8; ++v) {
      const int b = b0 + v + 8 * h;
      const float val = acc[v] + bs * myb;
      const unsigned int old = f3[b];
      if ((val >= vgate) && !((old >> l) & 1u)) {
        atomicOr(&f3[b], 1u << l);
        out[b * 10 + l] = tf;
      }
    }
  }
}

extern "C" void kernel_launch(void* const* d_in, const int* in_sizes, int n_in,
                              void* d_out, int out_size, void* d_ws, size_t ws_size,
                              hipStream_t stream) {
  (void)in_sizes; (void)n_in; (void)out_size; (void)ws_size;
  const float* img = (const float*)d_in[0];
  const float* w1  = (const float*)d_in[1];
  const float* b1  = (const float*)d_in[2];
  const float* w2  = (const float*)d_in[3];
  const float* b2  = (const float*)d_in[4];
  const float* fw  = (const float*)d_in[5];
  const float* fb  = (const float*)d_in[6];
  float* out = (float*)d_out;

  char* ws = (char*)d_ws;
  float*          st   = (float*)(ws + OFF_ST);
  unsigned short* f1   = (unsigned short*)(ws + OFF_F1);
  unsigned short* pm1  = (unsigned short*)(ws + OFF_PM1);
  unsigned int*   pm1t = (unsigned int*)(ws + OFF_PM1T);
  unsigned int*   f2   = (unsigned int*)(ws + OFF_F2);
  unsigned int*   pm2  = (unsigned int*)(ws + OFF_PM2);
  unsigned int*   f3   = (unsigned int*)(ws + OFF_F3);
  unsigned short* w2b  = (unsigned short*)(ws + OFF_W2B);
  unsigned short* fwb  = (unsigned short*)(ws + OFF_FWB);

  hipMemsetAsync(ws + OFF_F1, 0, STATE_BYTES, stream);
  k_init_out<<<(B_SZ * 10 + 255) / 256, 256, 0, stream>>>(out, B_SZ * 10);
  k_spike_times<<<(B_SZ * 784 + 255) / 256, 256, 0, stream>>>(img, st, B_SZ * 784);
  k_prep_w2<<<240, 256, 0, stream>>>(w2, w2b);
  k_prep_fw<<<192, 256, 0, stream>>>(fw, fwb);

  for (int t = 0; t < TW; ++t) {
    k_layer1<<<B_SZ, 256, 0, stream>>>(st, w1, b1, f1, pm1, pm1t, t);
    k_layer2<<<256, 256, 0, stream>>>(pm1t, (const uint4*)w2b, b2, f2, pm2, t);
    k_fc<<<32, 256, 0, stream>>>(pm2, (const uint4*)fwb, fb, f3, out, t);
  }
}
